// GPTNeoXGQA_74105365725689
// MI455X (gfx1250) — matrix operand order repaired and checked by reading, not compiled, not run on hardware
//
#include <hip/hip_runtime.h>
#include <hip/hip_bf16.h>

typedef __attribute__((ext_vector_type(16))) __bf16 v16bf;
typedef __attribute__((ext_vector_type(8)))  float  v8f;
typedef __attribute__((ext_vector_type(8)))  __bf16 bf16x8;

#define B_  2
#define S_  2048
#define HID_ 2048
#define H_  32
#define HK_ 8
#define D_  64
#define SCALE_ 0.125f

typedef unsigned long long u64;

union frag_u { v16bf v; bf16x8 h8[2]; };

// Async global->LDS 16-byte copy (ASYNCcnt path, CDNA5 §08).
#define ASYNC_B128(ldsoff, gaddr) \
  asm volatile("global_load_async_to_lds_b128 %0, %1, off" \
               :: "v"((unsigned)(ldsoff)), "v"((u64)(gaddr)) : "memory")
#define WAIT_ASYNC_0() asm volatile("s_wait_asynccnt 0x0" ::: "memory")
#define WAIT_ASYNC_2() asm volatile("s_wait_asynccnt 0x2" ::: "memory")
#define WAIT_ASYNC_5() asm volatile("s_wait_asynccnt 0x5" ::: "memory")
#define WAIT_DS_0()    asm volatile("s_wait_dscnt 0x0"    ::: "memory")

static __device__ inline unsigned ldsoff(const void* p) {
  return (unsigned)(u64)p;
}

static __device__ inline v8f zero8() {
  v8f z;
  #pragma unroll
  for (int e = 0; e < 8; ++e) z[e] = 0.0f;
  return z;
}

// A-fragment (16x32 bf16) from a row-major tile, row stride rs elements.
static __device__ inline v16bf load_a_frag(const __bf16* tile, int rs, int l16, int hf) {
  frag_u u;
  const __bf16* r = tile + (size_t)l16 * rs;
  u.h8[0] = *(const bf16x8*)(r + hf * 16);
  u.h8[1] = *(const bf16x8*)(r + hf * 16 + 8);
  return u.v;
}

// B-fragment (32x16 bf16): 16 contiguous K values per lane at p.
static __device__ inline v16bf load_b_frag(const __bf16* p) {
  frag_u u;
  u.h8[0] = *(const bf16x8*)(p);
  u.h8[1] = *(const bf16x8*)(p + 8);
  return u.v;
}

static __device__ inline v8f wmma_bf16(v16bf a, v16bf b, v8f c) {
  return __builtin_amdgcn_wmma_f32_16x16x32_bf16(false, a, false, b, (short)0, c,
                                                 false, false);
}

__global__ void f32_to_bf16_kernel(const float* __restrict__ in,
                                   __bf16* __restrict__ out, long n) {
  long i = (long)blockIdx.x * blockDim.x + threadIdx.x;
  long stride = (long)gridDim.x * blockDim.x;
  for (; i < n; i += stride) out[i] = (__bf16)in[i];
}

// out[N x K] = transpose(in[K x N]), fp32 -> bf16. 32x32 tiles via LDS.
__global__ __launch_bounds__(256) void transpose_f32_to_bf16_kernel(
    const float* __restrict__ in, __bf16* __restrict__ out, int K, int N) {
  __shared__ float tile[32][33];
  const int n0 = blockIdx.x * 32, k0 = blockIdx.y * 32;
  const int tx = threadIdx.x & 31, ty = threadIdx.x >> 5;  // ty: 0..7
  #pragma unroll
  for (int i = 0; i < 32; i += 8)
    tile[ty + i][tx] = in[(size_t)(k0 + ty + i) * N + n0 + tx];
  __syncthreads();
  #pragma unroll
  for (int i = 0; i < 32; i += 8)
    out[(size_t)(n0 + ty + i) * K + k0 + tx] = (__bf16)tile[tx][ty + i];
}

// C[M x N] = X[M x K] * Wt^T + bias, Wt stored transposed [N x K] (bf16).
// 256x64 block tile, 8 waves x 32 rows: each wave runs 8 WMMAs per K-step
// reusing 4 B-fragments across 2 A-fragments. Double-buffered async staging.
// vtrans: store bf16 output transposed per batch as [B][N][S_].
__global__ __launch_bounds__(256) void gemm_bias_kernel(
    const __bf16* __restrict__ X, const __bf16* __restrict__ Wt,
    const float* __restrict__ bias, __bf16* __restrict__ outb,
    float* __restrict__ outf, int M, int N, int K, int vtrans) {
  __shared__ alignas(16) __bf16 Xs[2][256][32];
  __shared__ alignas(16) __bf16 Ws[2][64][32];  // Ws[n][k]
  const int t = threadIdx.x;
  const int lane = t & 31, wave = t >> 5;
  const int hf = lane >> 4, l16 = lane & 15;
  const int n0 = blockIdx.x * 64;
  const int m0 = blockIdx.y * 256;

  auto issue_tile = [&](int k0, int buf) {
    #pragma unroll
    for (int j = 0; j < 4; ++j) {  // X tile: 256x32 = 1024 x 16B chunks
      int id = t + 256 * j;
      int r = id >> 2, c = id & 3;
      ASYNC_B128(ldsoff(&Xs[buf][r][c * 8]),
                 X + (size_t)(m0 + r) * K + k0 + c * 8);
    }
    {  // W tile: 64x32 = 256 x 16B chunks (K-contiguous rows)
      int n = t >> 2, c = t & 3;
      ASYNC_B128(ldsoff(&Ws[buf][n][c * 8]),
                 Wt + (size_t)(n0 + n) * K + k0 + c * 8);
    }
  };

  v8f acc0[4], acc1[4];
  #pragma unroll
  for (int nt = 0; nt < 4; ++nt) { acc0[nt] = zero8(); acc1[nt] = zero8(); }

  auto compute_tile = [&](int cur) {
    v16bf a0 = load_a_frag(&Xs[cur][wave * 32][0], 32, l16, hf);
    v16bf a1 = load_a_frag(&Xs[cur][wave * 32 + 16][0], 32, l16, hf);
    v16bf bf[4];
    #pragma unroll
    for (int nt = 0; nt < 4; ++nt)
      bf[nt] = load_b_frag(&Ws[cur][nt * 16 + l16][hf * 16]);
    #pragma unroll
    for (int nt = 0; nt < 4; ++nt) {
      acc0[nt] = wmma_bf16(a0, bf[nt], acc0[nt]);
      acc1[nt] = wmma_bf16(a1, bf[nt], acc1[nt]);
    }
  };

  const int nk = K / 32;
  issue_tile(0, 0);
  for (int kt = 0; kt < nk - 1; ++kt) {  // branch-free steady state
    const int cur = kt & 1;
    issue_tile((kt + 1) * 32, cur ^ 1);
    WAIT_ASYNC_5();  // 5 async ops per tile -> previous tile landed
    __syncthreads();
    compute_tile(cur);
    __syncthreads();
  }
  {  // peeled last iteration
    WAIT_ASYNC_0();
    __syncthreads();
    compute_tile((nk - 1) & 1);
  }

  #pragma unroll
  for (int strip = 0; strip < 2; ++strip) {
    #pragma unroll
    for (int nt = 0; nt < 4; ++nt) {
      #pragma unroll
      for (int i = 0; i < 8; ++i) {
        int row = m0 + wave * 32 + strip * 16 + i + hf * 8;
        int col = n0 + nt * 16 + l16;
        float v = (strip ? acc1[nt][i] : acc0[nt][i]) + bias[col];
        if (outf) {
          outf[(size_t)row * N + col] = v;
        } else if (vtrans) {
          int bb = row >> 11, s = row & (S_ - 1);
          outb[((size_t)bb * N + col) * S_ + s] = (__bf16)v;
        } else {
          outb[(size_t)row * N + col] = (__bf16)v;
        }
      }
    }
  }
}

// Flash attention: block = 8 waves x 16 q-rows; key tiles of 32; async
// double-buffered K/V staging. V comes pre-transposed: VvT[B][HK*D][S].
__global__ __launch_bounds__(256) void attn_kernel(
    const __bf16* __restrict__ Q, const __bf16* __restrict__ Kv,
    const __bf16* __restrict__ VvT, __bf16* __restrict__ O) {
  __shared__ alignas(16) __bf16 Kt[2][32][64];  // Kt[key][d]
  __shared__ alignas(16) __bf16 Vt[2][64][32];  // Vt[d][key]
  __shared__ alignas(16) __bf16 Ps[8][16][32];  // per-wave P scratch

  const int t = threadIdx.x;
  const int lane = t & 31, wave = t >> 5;
  const int hf = lane >> 4, l16 = lane & 15;
  const int qtile = blockIdx.x & 15;
  const int bh = blockIdx.x >> 4;
  const int h = bh & (H_ - 1);
  const int b = bh >> 5;
  const int hk = h >> 2;  // N_REP = 4
  const int qb = qtile * 128;
  const int q0 = qb + wave * 16;

  const __bf16* qptr   = Q + (size_t)b * S_ * (H_ * D_) + (size_t)h * D_;
  const __bf16* kbase  = Kv + (size_t)b * S_ * (HK_ * D_) + (size_t)hk * D_;
  const __bf16* vTbase = VvT + ((size_t)b * (HK_ * D_) + hk * D_) * S_;

  auto issue_tile = [&](int k0, int buf) {
    {  // K tile 32x64 = 256 x 16B chunks
      int kr = t >> 3, c = t & 7;
      ASYNC_B128(ldsoff(&Kt[buf][kr][c * 8]),
                 kbase + (size_t)(k0 + kr) * (HK_ * D_) + c * 8);
    }
    {  // V tile (transposed in memory) 64x32 = 256 x 16B chunks
      int d = t >> 2, c = t & 3;
      ASYNC_B128(ldsoff(&Vt[buf][d][c * 8]),
                 vTbase + (size_t)d * S_ + k0 + c * 8);
    }
  };

  // Preload Q fragments (16 rows x 64 as two 16x32 A-frags), direct global.
  v16bf qa[2];
  {
    const __bf16* r = qptr + (size_t)(q0 + l16) * (H_ * D_);
    #pragma unroll
    for (int c = 0; c < 2; ++c) {
      frag_u u;
      u.h8[0] = *(const bf16x8*)(r + c * 32 + hf * 16);
      u.h8[1] = *(const bf16x8*)(r + c * 32 + hf * 16 + 8);
      qa[c] = u.v;
    }
  }

  float mrow[8], lrow[8];
  v8f o[4];
  #pragma unroll
  for (int i = 0; i < 8; ++i) { mrow[i] = -1e30f; lrow[i] = 0.0f; }
  #pragma unroll
  for (int nt = 0; nt < 4; ++nt) o[nt] = zero8();

  auto compute_tile = [&](int kt) {
    const int k0 = kt * 32;
    const int cur = kt & 1;
    // S = Q(16x64) * K^T(64x32) -> two 16x16 fp32 frags.
    v8f s0 = zero8(), s1 = zero8();
    #pragma unroll
    for (int c = 0; c < 2; ++c) {
      v16bf b0 = load_b_frag(&Kt[cur][l16][c * 32 + hf * 16]);
      v16bf b1 = load_b_frag(&Kt[cur][16 + l16][c * 32 + hf * 16]);
      s0 = wmma_bf16(qa[c], b0, s0);
      s1 = wmma_bf16(qa[c], b1, s1);
    }

    // Causal mask + online softmax.
    float p0[8], p1[8];
    #pragma unroll
    for (int i = 0; i < 8; ++i) {
      int qrow = q0 + i + hf * 8;
      float v0 = s0[i] * SCALE_ + ((k0 + l16) > qrow ? -1e9f : 0.0f);
      float v1 = s1[i] * SCALE_ + ((k0 + 16 + l16) > qrow ? -1e9f : 0.0f);
      float rm = fmaxf(v0, v1);
      #pragma unroll
      for (int off = 1; off < 16; off <<= 1)
        rm = fmaxf(rm, __shfl_xor(rm, off, 32));
      float mnew = fmaxf(mrow[i], rm);
      float corr = __expf(mrow[i] - mnew);
      float e0 = __expf(v0 - mnew);
      float e1 = __expf(v1 - mnew);
      float rs = e0 + e1;
      #pragma unroll
      for (int off = 1; off < 16; off <<= 1)
        rs += __shfl_xor(rs, off, 32);
      lrow[i] = lrow[i] * corr + rs;
      mrow[i] = mnew;
      #pragma unroll
      for (int nt = 0; nt < 4; ++nt) o[nt][i] *= corr;
      p0[i] = e0;
      p1[i] = e1;
    }

    // P (C layout) -> wave-private LDS -> A-fragment layout.
    #pragma unroll
    for (int i = 0; i < 8; ++i) {
      Ps[wave][i + hf * 8][l16]      = (__bf16)p0[i];
      Ps[wave][i + hf * 8][16 + l16] = (__bf16)p1[i];
    }
    WAIT_DS_0();
    v16bf pa = load_a_frag(&Ps[wave][0][0], 32, l16, hf);

    // O += P(16x32) * V(32x64).
    #pragma unroll
    for (int nt = 0; nt < 4; ++nt) {
      v16bf vb = load_b_frag(&Vt[cur][nt * 16 + l16][hf * 16]);
      o[nt] = wmma_bf16(pa, vb, o[nt]);
    }
  };

  const int nkt = (qb + 128) / 32;  // causal: keys 0 .. qb+127 (nkt >= 4)
  issue_tile(0, 0);
  for (int kt = 0; kt < nkt - 1; ++kt) {  // branch-free steady state
    issue_tile((kt + 1) * 32, (kt & 1) ^ 1);
    WAIT_ASYNC_2();  // 2 async ops per tile
    __syncthreads();
    compute_tile(kt);
    __syncthreads();
  }
  {  // peeled last iteration
    WAIT_ASYNC_0();
    __syncthreads();
    compute_tile(nkt - 1);
  }

  // Normalize and store bf16 attn output [B*S, H*D].
  #pragma unroll
  for (int nt = 0; nt < 4; ++nt) {
    #pragma unroll
    for (int i = 0; i < 8; ++i) {
      int qrow = q0 + i + hf * 8;
      float val = o[nt][i] / lrow[i];
      O[((size_t)b * S_ + qrow) * (H_ * D_) + h * D_ + nt * 16 + l16] =
          (__bf16)val;
    }
  }
}

extern "C" void kernel_launch(void* const* d_in, const int* in_sizes, int n_in,
                              void* d_out, int out_size, void* d_ws,
                              size_t ws_size, hipStream_t stream) {
  const float* hs      = (const float*)d_in[0];
  // d_in[1] = attention_mask: pure causal, applied analytically in-kernel.
  const float* q_w     = (const float*)d_in[2];
  const float* q_b     = (const float*)d_in[3];
  const float* k_w     = (const float*)d_in[4];
  const float* k_b     = (const float*)d_in[5];
  const float* v_w     = (const float*)d_in[6];
  const float* v_b     = (const float*)d_in[7];
  const float* dense_w = (const float*)d_in[8];
  const float* dense_b = (const float*)d_in[9];
  float* out = (float*)d_out;

  const int M = B_ * S_;  // 4096

  char* cur = (char*)d_ws;
  auto alloc = [&](size_t bytes) -> void* {
    void* p = (void*)cur;
    cur += (bytes + 255) & ~(size_t)255;
    return p;
  };
  __bf16* hb    = (__bf16*)alloc((size_t)M * HID_ * 2);
  __bf16* qwt   = (__bf16*)alloc((size_t)HID_ * (H_ * D_) * 2);   // [N][K]
  __bf16* kwt   = (__bf16*)alloc((size_t)HID_ * (HK_ * D_) * 2);  // [N][K]
  __bf16* vwt   = (__bf16*)alloc((size_t)HID_ * (HK_ * D_) * 2);  // [N][K]
  __bf16* dwt   = (__bf16*)alloc((size_t)HID_ * HID_ * 2);        // [N][K]
  __bf16* qact  = (__bf16*)alloc((size_t)M * (H_ * D_) * 2);
  __bf16* kact  = (__bf16*)alloc((size_t)M * (HK_ * D_) * 2);
  __bf16* vactT = (__bf16*)alloc((size_t)M * (HK_ * D_) * 2);  // [B][HK*D][S]
  __bf16* attnb = (__bf16*)alloc((size_t)M * (H_ * D_) * 2);

  // 1) fp32 -> bf16 conversions; weights transposed once to [N][K].
  f32_to_bf16_kernel<<<2048, 256, 0, stream>>>(hs, hb, (long)M * HID_);
  transpose_f32_to_bf16_kernel<<<dim3((H_ * D_) / 32, HID_ / 32), 256, 0,
                                 stream>>>(q_w, qwt, HID_, H_ * D_);
  transpose_f32_to_bf16_kernel<<<dim3((HK_ * D_) / 32, HID_ / 32), 256, 0,
                                 stream>>>(k_w, kwt, HID_, HK_ * D_);
  transpose_f32_to_bf16_kernel<<<dim3((HK_ * D_) / 32, HID_ / 32), 256, 0,
                                 stream>>>(v_w, vwt, HID_, HK_ * D_);
  transpose_f32_to_bf16_kernel<<<dim3(HID_ / 32, HID_ / 32), 256, 0, stream>>>(
      dense_w, dwt, HID_, HID_);

  // 2) QKV projections (bf16 WMMA, fp32 accumulate). V written transposed.
  gemm_bias_kernel<<<dim3((H_ * D_) / 64, M / 256), 256, 0, stream>>>(
      hb, qwt, q_b, qact, nullptr, M, H_ * D_, HID_, 0);
  gemm_bias_kernel<<<dim3((HK_ * D_) / 64, M / 256), 256, 0, stream>>>(
      hb, kwt, k_b, kact, nullptr, M, HK_ * D_, HID_, 0);
  gemm_bias_kernel<<<dim3((HK_ * D_) / 64, M / 256), 256, 0, stream>>>(
      hb, vwt, v_b, vactT, nullptr, M, HK_ * D_, HID_, 1);

  // 3) Causal GQA flash attention.
  attn_kernel<<<B_ * H_ * (S_ / 128), 256, 0, stream>>>(qact, kact, vactT,
                                                        attnb);

  // 4) Output projection -> fp32 result.
  gemm_bias_kernel<<<dim3(HID_ / 64, M / 256), 256, 0, stream>>>(
      attnb, dwt, dense_b, nullptr, out, M, HID_, HID_, 0);
}
